// TemporalMambaEncoder_9320079033269
// MI455X (gfx1250) — matrix operand order repaired and checked by reading, not compiled, not run on hardware
//
#include <hip/hip_runtime.h>
#include <hip/hip_bf16.h>
#include <math.h>

// ---- problem constants (from reference) ----
#define BQ   4
#define TT   128
#define NN   22
#define DM   256
#define DI   512
#define DS   16
#define DTR  16
#define RR   (BQ * NN)        // 88 rows
#define MM   (RR * TT)        // 11264 GEMM rows

typedef _Float16 v16h __attribute__((ext_vector_type(16)));
typedef _Float16 v8h  __attribute__((ext_vector_type(8)));
typedef float    v8f  __attribute__((ext_vector_type(8)));

__device__ __forceinline__ v16h load_afrag(const _Float16* row, int k0, int grp)
{
    // 16-bit A 16x32 layout (ISA 7.12.2): lane-group g,
    // elements 0..7 -> K = k0+8g+0..7, elements 8..15 -> K = k0+8g+16..23
    v8h a0 = *(const v8h*)(row + k0 + grp * 8);
    v8h a1 = *(const v8h*)(row + k0 + grp * 8 + 16);
    return __builtin_shufflevector(a0, a1, 0,1,2,3,4,5,6,7,
                                           8,9,10,11,12,13,14,15);
}

__device__ __forceinline__ v16h load_bfrag(const _Float16* row, int k0, int grp)
{
    // 16-bit B 32x16 layout: lane-group g, elements 0..7 -> K = k0+8g+0..7, 8..15 -> K = k0+8g+16..23
    v8h b0 = *(const v8h*)(row + k0 + grp * 8);
    v8h b1 = *(const v8h*)(row + k0 + grp * 8 + 16);
    return __builtin_shufflevector(b0, b1, 0,1,2,3,4,5,6,7,
                                           8,9,10,11,12,13,14,15);
}

// =====================================================================
// Register-blocked wave-level WMMA GEMM on pre-converted f16 operands.
//   C[m,n] = sum_k A[m,k] * W[n,k]  (+bias[n])
// Each wave computes a 32 x (16*NT) output block: A fragments reused NT
// times, B fragments reused 2x -> 2*NT WMMAs per (4 + 2*NT) b128 loads.
// Requires K % 32 == 0, M % 32 == 0, N % (16*NT) == 0, rows 16B aligned.
// splitN: cols [0,splitN) -> C0, [splitN,N) -> C1 (f32 outputs).
// =====================================================================
template<int NT>
__global__ __launch_bounds__(32)
void gemm_wmma_kernel(const _Float16* __restrict__ A, int lda, int K,
                      const _Float16* __restrict__ W,
                      const float* __restrict__ bias,
                      float* __restrict__ C0, float* __restrict__ C1,
                      int N, int splitN, int accumulate)
{
    const int lane   = threadIdx.x & 31;
    const int grp    = lane >> 4;         // half-wave group (0 or 1)
    const int lane16 = lane & 15;

    const int mbase = blockIdx.x * 32;             // two 16-row tiles
    const int nbase = blockIdx.y * (16 * NT);      // NT 16-col tiles

    const _Float16* __restrict__ Arow0 = A + (size_t)(mbase + lane16) * lda;
    const _Float16* __restrict__ Arow1 = Arow0 + (size_t)16 * lda;
    const _Float16* Wrow[NT];
    #pragma unroll
    for (int nt = 0; nt < NT; ++nt)
        Wrow[nt] = W + (size_t)(nbase + nt * 16 + lane16) * K;

    v8f acc[2][NT];
    #pragma unroll
    for (int nt = 0; nt < NT; ++nt) { acc[0][nt] = {}; acc[1][nt] = {}; }

    for (int k0 = 0; k0 < K; k0 += 32) {
        v16h af0 = load_afrag(Arow0, k0, grp);
        v16h af1 = load_afrag(Arow1, k0, grp);
        v16h bf[NT];
        #pragma unroll
        for (int nt = 0; nt < NT; ++nt)
            bf[nt] = load_bfrag(Wrow[nt], k0, grp);
        #pragma unroll
        for (int nt = 0; nt < NT; ++nt) {
            acc[0][nt] = __builtin_amdgcn_wmma_f32_16x16x32_f16(
                false, af0, false, bf[nt], (short)0, acc[0][nt], false, false);
            acc[1][nt] = __builtin_amdgcn_wmma_f32_16x16x32_f16(
                false, af1, false, bf[nt], (short)0, acc[1][nt], false, false);
        }
    }

    // D layout: VGPR r holds M = tile + 8*grp + r, column n = tile + lane16
    #pragma unroll
    for (int mt = 0; mt < 2; ++mt) {
        #pragma unroll
        for (int nt = 0; nt < NT; ++nt) {
            const int nn = nbase + nt * 16 + lane16;
            const float bv = bias ? bias[nn] : 0.0f;
            #pragma unroll
            for (int r = 0; r < 8; ++r) {
                int mm = mbase + mt * 16 + grp * 8 + r;
                float v = acc[mt][nt][r] + bv;
                float* dst;
                if (nn < splitN) dst = &C0[(size_t)mm * splitN + nn];
                else             dst = &C1[(size_t)mm * (N - splitN) + (nn - splitN)];
                if (accumulate) *dst += v; else *dst = v;
            }
        }
    }
}

// ---- f32 -> f16 flat convert ----
__global__ void cvt_f16_kernel(const float* __restrict__ src,
                               _Float16* __restrict__ dst, size_t n)
{
    size_t i = (size_t)blockIdx.x * blockDim.x + threadIdx.x;
    if (i < n) dst[i] = (_Float16)src[i];
}

// ---- pad xdbl[:, :16] (f32, row stride 48) into (M,32) f16, cols>=16 zero ----
__global__ void pad_dtx_kernel(const float* __restrict__ xdb,
                               _Float16* __restrict__ dtx)
{
    size_t i = (size_t)blockIdx.x * blockDim.x + threadIdx.x;  // over M*32
    if (i >= (size_t)MM * 32) return;
    int c = (int)(i & 31);
    size_t m = i >> 5;
    dtx[i] = (c < DTR) ? (_Float16)xdb[m * (DTR + 2 * DS) + c] : (_Float16)0.0f;
}

// ---- pad dt_w (512,16) f32 into (512,32) f16, cols>=16 zero ----
__global__ void pad_dtw_kernel(const float* __restrict__ w,
                               _Float16* __restrict__ wp)
{
    size_t i = (size_t)blockIdx.x * blockDim.x + threadIdx.x;  // over 512*32
    if (i >= (size_t)DI * 32) return;
    int c = (int)(i & 31);
    size_t n = i >> 5;
    wp[i] = (c < DTR) ? (_Float16)w[n * DTR + c] : (_Float16)0.0f;
}

// ---- S[r=(b*NN+n), t, :] = x[b, flip? T-1-t : t, n, :] ----
__global__ void reorder_in_kernel(const float* __restrict__ x,
                                  float* __restrict__ S, int flip)
{
    size_t i = (size_t)blockIdx.x * blockDim.x + threadIdx.x;
    if (i >= (size_t)RR * TT * DM) return;
    int d = (int)(i % DM);
    size_t j = i / DM;
    int t = (int)(j % TT);
    int r = (int)(j / TT);
    int b = r / NN, n = r % NN;
    int ts = flip ? (TT - 1 - t) : t;
    S[i] = x[(((size_t)b * TT + ts) * NN + n) * DM + d];
}

// ---- out[b,t,n,:] (+)= 0.5 * S[b*NN+n, flip? T-1-t : t, :] ----
__global__ void writer_kernel(const float* __restrict__ S,
                              float* __restrict__ out, int flip, int init)
{
    size_t i = (size_t)blockIdx.x * blockDim.x + threadIdx.x;
    if (i >= (size_t)BQ * TT * NN * DM) return;
    int d = (int)(i % DM);
    size_t j = i / DM;
    int n = (int)(j % NN); j /= NN;
    int t = (int)(j % TT);
    int b = (int)(j / TT);
    int ts = flip ? (TT - 1 - t) : t;
    float v = 0.5f * S[(((size_t)(b * NN + n)) * TT + ts) * DM + d];
    if (init) out[i] = v; else out[i] += v;
}

// ---- depthwise causal conv (width 4) + bias + silu; emits f32 and f16 ----
__global__ void conv_silu_kernel(const float* __restrict__ u,
                                 const float* __restrict__ cw,
                                 const float* __restrict__ cb,
                                 float* __restrict__ uc,
                                 _Float16* __restrict__ uc16)
{
    size_t i = (size_t)blockIdx.x * blockDim.x + threadIdx.x;
    if (i >= (size_t)RR * TT * DI) return;
    int c = (int)(i % DI);
    size_t j = i / DI;
    int t = (int)(j % TT);
    int r = (int)(j / TT);
    float acc = cb[c];
    #pragma unroll
    for (int k = 0; k < 4; ++k) {
        int tt = t + k - 3;
        if (tt >= 0)
            acc += u[((size_t)r * TT + tt) * DI + c] * cw[c * 4 + k];
    }
    float s = acc / (1.0f + expf(-acc));   // silu
    uc[i] = s;
    uc16[i] = (_Float16)s;
}

__global__ void softplus_kernel(float* __restrict__ v)
{
    size_t i = (size_t)blockIdx.x * blockDim.x + threadIdx.x;
    if (i >= (size_t)MM * DI) return;
    float x = v[i];
    v[i] = (x > 20.0f) ? x : log1pf(expf(x));
}

// =====================================================================
// Selective scan: one block per row r, one thread per channel c.
// =====================================================================
__global__ __launch_bounds__(DI)
void scan_kernel(const float* __restrict__ dt,
                 const float* __restrict__ xdbl,   // (M,48): [16:32)=B, [32:48)=C
                 const float* __restrict__ uc,
                 const float* __restrict__ a_log,  // (DI, DS)
                 float* __restrict__ ys)
{
    const int r = blockIdx.x;
    const int c = threadIdx.x;
    __shared__ float sB[DS];
    __shared__ float sC[DS];

    float A[DS];
    #pragma unroll
    for (int s = 0; s < DS; ++s) A[s] = -expf(a_log[c * DS + s]);
    float h[DS];
    #pragma unroll
    for (int s = 0; s < DS; ++s) h[s] = 0.0f;

    for (int t = 0; t < TT; ++t) {
        const float* xrow = xdbl + ((size_t)r * TT + t) * (DTR + 2 * DS);
        if (c < 2 * DS) {
            if (c < DS) sB[c]      = xrow[DTR + c];
            else        sC[c - DS] = xrow[DTR + DS + (c - DS)];
        }
        __syncthreads();
        size_t idx = ((size_t)r * TT + t) * DI + c;
        float dtv = dt[idx];
        float uv  = uc[idx];
        float du  = dtv * uv;
        float y = 0.0f;
        #pragma unroll
        for (int s = 0; s < DS; ++s) {
            float dA = expf(dtv * A[s]);
            h[s] = dA * h[s] + du * sB[s];
            y += h[s] * sC[s];
        }
        ys[idx] = y;
        __syncthreads();
    }
}

// ---- y16 = (ys + uc * D_skip[c]) * silu(z), written as f16 GEMM operand ----
__global__ void gate_kernel(const float* __restrict__ ys,
                            const float* __restrict__ uc,
                            const float* __restrict__ z,
                            const float* __restrict__ dskip,
                            _Float16* __restrict__ y16)
{
    size_t i = (size_t)blockIdx.x * blockDim.x + threadIdx.x;
    if (i >= (size_t)MM * DI) return;
    int c = (int)(i % DI);
    float zv = z[i];
    float sz = zv / (1.0f + expf(-zv));
    y16[i] = (_Float16)((ys[i] + uc[i] * dskip[c]) * sz);
}

// =====================================================================
extern "C" void kernel_launch(void* const* d_in, const int* in_sizes, int n_in,
                              void* d_out, int out_size, void* d_ws, size_t ws_size,
                              hipStream_t stream)
{
    const float* x      = (const float*)d_in[0];
    const float* in_w   = (const float*)d_in[1];
    const float* conv_w = (const float*)d_in[2];
    const float* conv_b = (const float*)d_in[3];
    const float* xp_w   = (const float*)d_in[4];
    const float* dt_w   = (const float*)d_in[5];
    const float* dt_b   = (const float*)d_in[6];
    const float* a_log  = (const float*)d_in[7];
    const float* d_skip = (const float*)d_in[8];
    const float* out_w  = (const float*)d_in[9];
    float* out = (float*)d_out;

    // ---- workspace carve-up ----
    float* S    = (float*)d_ws;                    // (M, 256)
    float* u    = S    + (size_t)MM * DM;          // (M, 512)
    float* z    = u    + (size_t)MM * DI;
    float* uc   = z    + (size_t)MM * DI;
    float* dtb  = uc   + (size_t)MM * DI;
    float* ys   = dtb  + (size_t)MM * DI;
    float* xdb  = ys   + (size_t)MM * DI;          // (M, 48)
    float* fend = xdb  + (size_t)MM * (DTR + 2 * DS);
    _Float16* S16  = (_Float16*)fend;              // (M, 256)
    _Float16* uc16 = S16  + (size_t)MM * DM;       // (M, 512)
    _Float16* y16  = uc16 + (size_t)MM * DI;       // (M, 512)
    _Float16* dtx  = y16  + (size_t)MM * DI;       // (M, 32) padded
    _Float16* wbuf = dtx  + (size_t)MM * 32;       // up to 1024*256 halves

    const int EW = 256;
    const size_t nS  = (size_t)MM * DM;
    const size_t nDI = (size_t)MM * DI;
    #define GRID(n) ((unsigned)(((n) + EW - 1) / EW))

    for (int dir = 0; dir < 2; ++dir) {
        reorder_in_kernel<<<GRID(nS), EW, 0, stream>>>(x, S, dir);

        for (int l = 0; l < 2; ++l) {
            const int off = dir * 2 + l;
            const float* p_inw  = in_w   + (size_t)off * 2 * DI * DM;
            const float* p_cw   = conv_w + (size_t)off * DI * 4;
            const float* p_cb   = conv_b + (size_t)off * DI;
            const float* p_xpw  = xp_w   + (size_t)off * (DTR + 2 * DS) * DI;
            const float* p_dtw  = dt_w   + (size_t)off * DI * DTR;
            const float* p_dtb  = dt_b   + (size_t)off * DI;
            const float* p_alog = a_log  + (size_t)off * DI * DS;
            const float* p_dsk  = d_skip + (size_t)off * DI;
            const float* p_outw = out_w  + (size_t)off * DM * DI;

            // ---- in_proj: (M,256) x (1024,256)^T -> u | z ----
            cvt_f16_kernel<<<GRID(nS), EW, 0, stream>>>(S, S16, nS);
            cvt_f16_kernel<<<GRID((size_t)2 * DI * DM), EW, 0, stream>>>(
                p_inw, wbuf, (size_t)2 * DI * DM);
            gemm_wmma_kernel<4><<<dim3(MM / 32, (2 * DI) / 64), 32, 0, stream>>>(
                S16, DM, DM, wbuf, nullptr, u, z, 2 * DI, DI, 0);

            conv_silu_kernel<<<GRID(nDI), EW, 0, stream>>>(u, p_cw, p_cb, uc, uc16);

            // ---- x_proj: (M,512) x (48,512)^T -> xdbl (N=48 -> NT=3) ----
            cvt_f16_kernel<<<GRID((size_t)(DTR + 2 * DS) * DI), EW, 0, stream>>>(
                p_xpw, wbuf, (size_t)(DTR + 2 * DS) * DI);
            gemm_wmma_kernel<3><<<dim3(MM / 32, 1), 32, 0, stream>>>(
                uc16, DI, DI, wbuf, nullptr, xdb, nullptr,
                DTR + 2 * DS, DTR + 2 * DS, 0);

            // ---- dt_proj: K=16 zero-padded to 32 on both operands ----
            pad_dtx_kernel<<<GRID((size_t)MM * 32), EW, 0, stream>>>(xdb, dtx);
            pad_dtw_kernel<<<GRID((size_t)DI * 32), EW, 0, stream>>>(p_dtw, wbuf);
            gemm_wmma_kernel<4><<<dim3(MM / 32, DI / 64), 32, 0, stream>>>(
                dtx, 32, 32, wbuf, p_dtb, dtb, nullptr, DI, DI, 0);

            softplus_kernel<<<GRID(nDI), EW, 0, stream>>>(dtb);

            scan_kernel<<<RR, DI, 0, stream>>>(dtb, xdb, uc, p_alog, ys);

            gate_kernel<<<GRID(nDI), EW, 0, stream>>>(ys, uc, z, p_dsk, y16);

            // ---- out_proj: (M,512) x (256,512)^T accumulated into S ----
            cvt_f16_kernel<<<GRID((size_t)DM * DI), EW, 0, stream>>>(
                p_outw, wbuf, (size_t)DM * DI);
            gemm_wmma_kernel<4><<<dim3(MM / 32, DM / 64), 32, 0, stream>>>(
                y16, DI, DI, wbuf, nullptr, S, nullptr, DM, DM, 1);
        }

        writer_kernel<<<GRID(nS), EW, 0, stream>>>(S, out, dir, dir == 0 ? 1 : 0);
    }
    #undef GRID
}
